// NonLocalBlock_33681133535995
// MI455X (gfx1250) — matrix operand order repaired and checked by reading, not compiled, not run on hardware
//
#include <hip/hip_runtime.h>
#include <hip/hip_bf16.h>

// Problem constants
#define B_   4
#define C_   1024
#define CI_  512
#define T_   16
#define H_   28
#define W_   28
#define N_   (T_*H_*W_)      // 12544 tokens per batch
#define NK_  (T_*14*14)      // 3136 pooled tokens per batch
#define BN_EPS 1e-5f

typedef __bf16 bf16;
typedef __attribute__((ext_vector_type(16))) bf16  v16bf;
typedef __attribute__((ext_vector_type(8)))  bf16  v8bf;
typedef __attribute__((ext_vector_type(8)))  float v8f;

// ---------------------------------------------------------------------------
// f32 -> bf16 elementwise convert (weights)
// ---------------------------------------------------------------------------
__global__ __launch_bounds__(256) void convert_f32_bf16(const float* __restrict__ in,
                                                        bf16* __restrict__ out, int n) {
  int i = blockIdx.x * 256 + threadIdx.x;
  if (i < n) out[i] = (bf16)in[i];
}

// ---------------------------------------------------------------------------
// Tiled transpose: x (B, C, N) f32 -> Xt (B, N, C) bf16   (token-major)
// block (32,8), grid (N/32, C/32, B)
// ---------------------------------------------------------------------------
__global__ __launch_bounds__(256) void transpose_to_tokens(const float* __restrict__ x,
                                                           bf16* __restrict__ Xt) {
  __shared__ float tile[32][33];
  const int b  = blockIdx.z;
  const int n0 = blockIdx.x * 32;
  const int c0 = blockIdx.y * 32;
  const float* xb = x  + (long long)b * C_ * N_;
  bf16*        xt = Xt + (long long)b * N_ * C_;
  const int tx = threadIdx.x, ty = threadIdx.y;
#pragma unroll
  for (int i = 0; i < 32; i += 8)
    tile[ty + i][tx] = xb[(long long)(c0 + ty + i) * N_ + (n0 + tx)];
  __syncthreads();
#pragma unroll
  for (int i = 0; i < 32; i += 8)
    xt[(long long)(n0 + ty + i) * C_ + (c0 + tx)] = (bf16)tile[tx][ty + i];
}

// ---------------------------------------------------------------------------
// MaxPool (1,2,2) in token-major space: Xt (B,N,C) -> Xpt (B,Nk,C), coalesced
// grid (C/256, Nk, B), block 256
// ---------------------------------------------------------------------------
__global__ __launch_bounds__(256) void pool_tokens(const bf16* __restrict__ Xt,
                                                   bf16* __restrict__ Xpt) {
  const int b  = blockIdx.z;
  const int nk = blockIdx.y;
  const int c  = blockIdx.x * 256 + threadIdx.x;
  const int t  = nk / 196;
  const int rm = nk % 196;
  const int h2 = rm / 14;
  const int w2 = rm % 14;
  const int n00 = t * 784 + h2 * 56 + w2 * 2;
  const bf16* base = Xt + (long long)b * N_ * C_;
  float v0 = (float)base[(long long)(n00     ) * C_ + c];
  float v1 = (float)base[(long long)(n00 +  1) * C_ + c];
  float v2 = (float)base[(long long)(n00 + 28) * C_ + c];
  float v3 = (float)base[(long long)(n00 + 29) * C_ + c];
  float v = fmaxf(fmaxf(v0, v1), fmaxf(v2, v3));
  Xpt[((long long)b * NK_ + nk) * C_ + c] = (bf16)v;
}

// ---------------------------------------------------------------------------
// NT GEMM: D(M,Nc) = alpha * A(M,K) x B(Nc,K)^T  (+ bias), bf16 in, f32 acc
// Fragment layouts used here (contiguous k; same order for A and B):
//   A lane: row = lane%16, half = lane/16; K = {16h..16h+15}
//   B lane: col = lane%16, half = lane/16; K = {16h..16h+15}
// block = 64 threads (2 waves); each wave: 32 rows x 64 cols (8 wmma tiles)
// Software-pipelined: iteration k+1's fragments are loaded before iteration
// k's WMMAs so s_wait_loadcnt only has to cover the older load group.
// grid (Nc/64, M/64, batches).  bias_mode: 0 none, 1 per-column, 2 per-row
// ---------------------------------------------------------------------------
__device__ __forceinline__ v16bf load_a_frag(const bf16* __restrict__ row, int k0, int half) {
  v8bf lo = *(const v8bf*)(row + k0 +     half * 16);
  v8bf hi = *(const v8bf*)(row + k0 + 8 + half * 16);
  v16bf r;
#pragma unroll
  for (int i = 0; i < 8; ++i) { r[i] = lo[i]; r[i + 8] = hi[i]; }
  return r;
}

#define WMMA_BF16(a, b, c) \
  __builtin_amdgcn_wmma_f32_16x16x32_bf16(false, (a), false, (b), (short)0, (c), false, false)

__global__ __launch_bounds__(64) void gemm_nt_bf16(
    const bf16* __restrict__ A, const bf16* __restrict__ Bm,
    const float* __restrict__ bias, int bias_mode, float alpha,
    float* __restrict__ outF, bf16* __restrict__ outH,
    int M, int Nc, int K,
    long long aBatch, long long bBatch, long long oBatch) {
  const int wave = threadIdx.x >> 5;
  const int lane = threadIdx.x & 31;
  const int r16  = lane & 15;
  const int half = lane >> 4;
  const long long bz = blockIdx.z;
  const int m0 = blockIdx.y * 64 + wave * 32;
  const int n0 = blockIdx.x * 64;

  const bf16* Ab = A  + bz * aBatch;
  const bf16* Bb = Bm + bz * bBatch;

  const bf16* aRow0 = Ab + (long long)(m0      + r16) * K;
  const bf16* aRow1 = Ab + (long long)(m0 + 16 + r16) * K;
  const bf16* bRow0 = Bb + (long long)(n0      + r16) * K;
  const bf16* bRow1 = Bb + (long long)(n0 + 16 + r16) * K;
  const bf16* bRow2 = Bb + (long long)(n0 + 32 + r16) * K;
  const bf16* bRow3 = Bb + (long long)(n0 + 48 + r16) * K;

  v8f acc[2][4];
#pragma unroll
  for (int tm = 0; tm < 2; ++tm)
#pragma unroll
    for (int t = 0; t < 4; ++t)
#pragma unroll
      for (int j = 0; j < 8; ++j) acc[tm][t][j] = 0.0f;

  // Prologue: fragments for k0 = 0
  v16bf a0 = load_a_frag(aRow0, 0, half);
  v16bf a1 = load_a_frag(aRow1, 0, half);
  v16bf b0 = *(const v16bf*)(bRow0 + 16 * half);
  v16bf b1 = *(const v16bf*)(bRow1 + 16 * half);
  v16bf b2 = *(const v16bf*)(bRow2 + 16 * half);
  v16bf b3 = *(const v16bf*)(bRow3 + 16 * half);

  for (int k0 = 0; k0 + 32 < K; k0 += 32) {
    const int kn = k0 + 32;
    // Load next iteration's fragments first (overlaps with WMMAs below)
    v16bf na0 = load_a_frag(aRow0, kn, half);
    v16bf na1 = load_a_frag(aRow1, kn, half);
    v16bf nb0 = *(const v16bf*)(bRow0 + kn + 16 * half);
    v16bf nb1 = *(const v16bf*)(bRow1 + kn + 16 * half);
    v16bf nb2 = *(const v16bf*)(bRow2 + kn + 16 * half);
    v16bf nb3 = *(const v16bf*)(bRow3 + kn + 16 * half);
    __builtin_prefetch(aRow0 + kn + 32, 0, 1);

    acc[0][0] = WMMA_BF16(a0, b0, acc[0][0]);
    acc[1][0] = WMMA_BF16(a1, b0, acc[1][0]);
    acc[0][1] = WMMA_BF16(a0, b1, acc[0][1]);
    acc[1][1] = WMMA_BF16(a1, b1, acc[1][1]);
    acc[0][2] = WMMA_BF16(a0, b2, acc[0][2]);
    acc[1][2] = WMMA_BF16(a1, b2, acc[1][2]);
    acc[0][3] = WMMA_BF16(a0, b3, acc[0][3]);
    acc[1][3] = WMMA_BF16(a1, b3, acc[1][3]);

    a0 = na0; a1 = na1; b0 = nb0; b1 = nb1; b2 = nb2; b3 = nb3;
  }

  // Epilogue k-step (no next loads)
  acc[0][0] = WMMA_BF16(a0, b0, acc[0][0]);
  acc[1][0] = WMMA_BF16(a1, b0, acc[1][0]);
  acc[0][1] = WMMA_BF16(a0, b1, acc[0][1]);
  acc[1][1] = WMMA_BF16(a1, b1, acc[1][1]);
  acc[0][2] = WMMA_BF16(a0, b2, acc[0][2]);
  acc[1][2] = WMMA_BF16(a1, b2, acc[1][2]);
  acc[0][3] = WMMA_BF16(a0, b3, acc[0][3]);
  acc[1][3] = WMMA_BF16(a1, b3, acc[1][3]);

  // Store: D lane layout -> row = m0 + 16*tm + r + 8*half, col = n0 + 16t + lane%16
  float* oF = outF ? outF + bz * oBatch : nullptr;
  bf16*  oH = outH ? outH + bz * oBatch : nullptr;
#pragma unroll
  for (int tm = 0; tm < 2; ++tm) {
#pragma unroll
    for (int t = 0; t < 4; ++t) {
      const int col = n0 + 16 * t + r16;
      const float cb = (bias_mode == 1) ? bias[col] : 0.0f;
#pragma unroll
      for (int r = 0; r < 8; ++r) {
        const int row = m0 + 16 * tm + r + 8 * half;
        float v = acc[tm][t][r] * alpha + cb;
        if (bias_mode == 2) v += bias[row];
        const long long idx = (long long)row * Nc + col;
        if (oF) oF[idx] = v;
        else    oH[idx] = (bf16)v;
      }
    }
  }
}

// ---------------------------------------------------------------------------
// Row softmax: f (rows, Nk) f32 -> attn (rows, Nk) bf16. One block per row.
// ---------------------------------------------------------------------------
__global__ __launch_bounds__(256) void softmax_rows(const float* __restrict__ f,
                                                    bf16* __restrict__ attn, int Nk) {
  const long long row = blockIdx.x;
  const float* fr = f    + row * Nk;
  bf16*        ar = attn + row * Nk;
  __shared__ float red[8];
  const int tid = threadIdx.x;

  float m = -3.402823466e38f;
  for (int i = tid; i < Nk; i += 256) m = fmaxf(m, fr[i]);
#pragma unroll
  for (int o = 16; o > 0; o >>= 1) m = fmaxf(m, __shfl_xor(m, o, 32));
  if ((tid & 31) == 0) red[tid >> 5] = m;
  __syncthreads();
  float M = -3.402823466e38f;
#pragma unroll
  for (int j = 0; j < 8; ++j) M = fmaxf(M, red[j]);
  __syncthreads();

  float s = 0.0f;
  for (int i = tid; i < Nk; i += 256) s += __expf(fr[i] - M);
#pragma unroll
  for (int o = 16; o > 0; o >>= 1) s += __shfl_xor(s, o, 32);
  if ((tid & 31) == 0) red[tid >> 5] = s;
  __syncthreads();
  float S = 0.0f;
#pragma unroll
  for (int j = 0; j < 8; ++j) S += red[j];
  const float inv = 1.0f / S;

  for (int i = tid; i < Nk; i += 256) ar[i] = (bf16)(__expf(fr[i] - M) * inv);
}

// ---------------------------------------------------------------------------
// BatchNorm pass 1: partial per-channel sums over row chunks.
// wy is (B*N, C). grid (C/256, nParts), block 256. rowsPerPart divides B*N.
// ---------------------------------------------------------------------------
__global__ __launch_bounds__(256) void bn_partial(const float* __restrict__ wy,
                                                  float* __restrict__ psum,
                                                  float* __restrict__ psq,
                                                  int rowsPerPart) {
  const int c  = blockIdx.x * 256 + threadIdx.x;
  const int r0 = blockIdx.y * rowsPerPart;
  float s = 0.0f, q = 0.0f;
  for (int r = r0; r < r0 + rowsPerPart; ++r) {
    float v = wy[(long long)r * C_ + c];
    s += v; q += v * v;
  }
  psum[blockIdx.y * C_ + c] = s;
  psq [blockIdx.y * C_ + c] = q;
}

// BatchNorm pass 2: finalize scale/shift per channel.
__global__ __launch_bounds__(256) void bn_stats(const float* __restrict__ psum,
                                                const float* __restrict__ psq,
                                                const float* __restrict__ gamma,
                                                const float* __restrict__ beta,
                                                float* __restrict__ scale,
                                                float* __restrict__ shift,
                                                int nParts, float invCount) {
  const int c = blockIdx.x * 256 + threadIdx.x;
  float s = 0.0f, q = 0.0f;
  for (int p = 0; p < nParts; ++p) { s += psum[p * C_ + c]; q += psq[p * C_ + c]; }
  const float mean = s * invCount;
  const float var  = q * invCount - mean * mean;
  const float sc   = gamma[c] * rsqrtf(var + BN_EPS);
  scale[c] = sc;
  shift[c] = beta[c] - mean * sc;
}

// ---------------------------------------------------------------------------
// Final: BN apply + transpose back (B,N,C)->(B,C,N) + residual add of x.
// block (32,8), grid (N/32, C/32, B)
// ---------------------------------------------------------------------------
__global__ __launch_bounds__(256) void bn_apply_transpose(const float* __restrict__ wy,
                                                          const float* __restrict__ x,
                                                          const float* __restrict__ scale,
                                                          const float* __restrict__ shift,
                                                          float* __restrict__ out) {
  __shared__ float tile[32][33];
  const int b  = blockIdx.z;
  const int n0 = blockIdx.x * 32;
  const int c0 = blockIdx.y * 32;
  const float* wyb = wy + (long long)b * N_ * C_;
  const float* xb  = x  + (long long)b * C_ * N_;
  float*       ob  = out + (long long)b * C_ * N_;
  const int tx = threadIdx.x, ty = threadIdx.y;
#pragma unroll
  for (int i = 0; i < 32; i += 8) {
    const int c = c0 + tx, n = n0 + ty + i;
    tile[ty + i][tx] = wyb[(long long)n * C_ + c] * scale[c] + shift[c];
  }
  __syncthreads();
#pragma unroll
  for (int i = 0; i < 32; i += 8) {
    const int c = c0 + ty + i, n = n0 + tx;
    ob[(long long)c * N_ + n] = tile[tx][ty + i] + xb[(long long)c * N_ + n];
  }
}

// ---------------------------------------------------------------------------
extern "C" void kernel_launch(void* const* d_in, const int* in_sizes, int n_in,
                              void* d_out, int out_size, void* d_ws, size_t ws_size,
                              hipStream_t stream) {
  const float* x       = (const float*)d_in[0];
  const float* theta_w = (const float*)d_in[1];
  const float* theta_b = (const float*)d_in[2];
  const float* phi_w   = (const float*)d_in[3];
  const float* phi_b   = (const float*)d_in[4];
  const float* g_w     = (const float*)d_in[5];
  const float* g_b     = (const float*)d_in[6];
  const float* w_w     = (const float*)d_in[7];
  const float* w_b     = (const float*)d_in[8];
  const float* bn_g    = (const float*)d_in[9];
  const float* bn_b    = (const float*)d_in[10];
  float* out = (float*)d_out;

  // workspace carving (256B aligned)
  char* ws = (char*)d_ws;
  size_t off = 0;
  auto alloc = [&](size_t bytes) -> void* {
    void* p = ws + off;
    off = (off + bytes + 255) & ~(size_t)255;
    return p;
  };
  const int WCNT = CI_ * C_;                                    // 524288 (all 4 weights)
  bf16*  Xt    = (bf16*) alloc((size_t)B_ * N_  * C_  * 2);     // token-major x
  bf16*  Xpt   = (bf16*) alloc((size_t)B_ * NK_ * C_  * 2);     // pooled token-major
  bf16*  thW   = (bf16*) alloc((size_t)WCNT * 2);
  bf16*  phW   = (bf16*) alloc((size_t)WCNT * 2);
  bf16*  gW    = (bf16*) alloc((size_t)WCNT * 2);
  bf16*  wW    = (bf16*) alloc((size_t)WCNT * 2);
  bf16*  theta = (bf16*) alloc((size_t)B_ * N_  * CI_ * 2);     // (N, CI)
  bf16*  phi   = (bf16*) alloc((size_t)B_ * NK_ * CI_ * 2);     // (Nk, CI)
  bf16*  gmat  = (bf16*) alloc((size_t)B_ * CI_ * NK_ * 2);     // (CI, Nk)
  float* fbuf  = (float*)alloc((size_t)N_ * NK_ * 4);           // per-batch scratch
  bf16*  attn  = (bf16*) alloc((size_t)N_ * NK_ * 2);           // per-batch scratch
  bf16*  ybuf  = (bf16*) alloc((size_t)B_ * N_  * CI_ * 2);     // (N, CI)
  float* wy    = (float*)alloc((size_t)B_ * N_  * C_  * 4);     // (N, C) f32
  const int NPARTS = 128;
  float* psum  = (float*)alloc((size_t)NPARTS * C_ * 4);
  float* psq   = (float*)alloc((size_t)NPARTS * C_ * 4);
  float* scale = (float*)alloc((size_t)C_ * 4);
  float* shift = (float*)alloc((size_t)C_ * 4);
  (void)ws_size; (void)in_sizes; (void)n_in; (void)out_size;

  // 1) weights -> bf16
  convert_f32_bf16<<<dim3(WCNT / 256), dim3(256), 0, stream>>>(theta_w, thW, WCNT);
  convert_f32_bf16<<<dim3(WCNT / 256), dim3(256), 0, stream>>>(phi_w,   phW, WCNT);
  convert_f32_bf16<<<dim3(WCNT / 256), dim3(256), 0, stream>>>(g_w,     gW,  WCNT);
  convert_f32_bf16<<<dim3(WCNT / 256), dim3(256), 0, stream>>>(w_w,     wW,  WCNT);

  // 2) x -> token-major bf16; 3) maxpool in token space
  transpose_to_tokens<<<dim3(N_ / 32, C_ / 32, B_), dim3(32, 8), 0, stream>>>(x, Xt);
  pool_tokens<<<dim3(C_ / 256, NK_, B_), dim3(256), 0, stream>>>(Xt, Xpt);

  const long long sXt = (long long)N_ * C_, sXpt = (long long)NK_ * C_;
  const long long sTh = (long long)N_ * CI_, sPh = (long long)NK_ * CI_;
  const long long sG  = (long long)CI_ * NK_, sWy = (long long)N_ * C_;

  // 4) theta = Xt . thW^T + b    (N x CI, K=C)
  gemm_nt_bf16<<<dim3(CI_ / 64, N_ / 64, B_), dim3(64), 0, stream>>>(
      Xt, thW, theta_b, 1, 1.0f, nullptr, theta, N_, CI_, C_, sXt, 0, sTh);
  // 5) phi = Xpt . phW^T + b     (Nk x CI, K=C)
  gemm_nt_bf16<<<dim3(CI_ / 64, NK_ / 64, B_), dim3(64), 0, stream>>>(
      Xpt, phW, phi_b, 1, 1.0f, nullptr, phi, NK_, CI_, C_, sXpt, 0, sPh);
  // 6) gmat = gW . Xpt^T + b(row) (CI x Nk, K=C)
  gemm_nt_bf16<<<dim3(NK_ / 64, CI_ / 64, B_), dim3(64), 0, stream>>>(
      gW, Xpt, g_b, 2, 1.0f, nullptr, gmat, CI_, NK_, C_, 0, sXpt, sG);

  // 7) per-batch attention (stream-serialized; reuses fbuf/attn scratch)
  const float alpha = 0.04419417382415922f;  // CI^-0.5
  for (int b = 0; b < B_; ++b) {
    gemm_nt_bf16<<<dim3(NK_ / 64, N_ / 64, 1), dim3(64), 0, stream>>>(
        theta + (long long)b * sTh, phi + (long long)b * sPh,
        nullptr, 0, alpha, fbuf, nullptr, N_, NK_, CI_, 0, 0, 0);
    softmax_rows<<<dim3(N_), dim3(256), 0, stream>>>(fbuf, attn, NK_);
    gemm_nt_bf16<<<dim3(CI_ / 64, N_ / 64, 1), dim3(64), 0, stream>>>(
        attn, gmat + (long long)b * sG,
        nullptr, 0, 1.0f, nullptr, ybuf + (long long)b * sTh, N_, CI_, NK_, 0, 0, 0);
  }

  // 8) wy = y . wW^T + b         (N x C, K=CI), f32 out for BN stats
  gemm_nt_bf16<<<dim3(C_ / 64, N_ / 64, B_), dim3(64), 0, stream>>>(
      ybuf, wW, w_b, 1, 1.0f, wy, nullptr, N_, C_, CI_, sTh, 0, sWy);

  // 9) BatchNorm (training-mode batch stats) + residual + transpose back
  const int totalRows = B_ * N_;                 // 50176
  const int rowsPerPart = totalRows / NPARTS;    // 392, exact
  bn_partial<<<dim3(C_ / 256, NPARTS), dim3(256), 0, stream>>>(wy, psum, psq, rowsPerPart);
  bn_stats<<<dim3(C_ / 256), dim3(256), 0, stream>>>(psum, psq, bn_g, bn_b, scale, shift,
                                                     NPARTS, 1.0f / (float)totalRows);
  bn_apply_transpose<<<dim3(N_ / 32, C_ / 32, B_), dim3(32, 8), 0, stream>>>(
      wy, x, scale, shift, out);
}
